// SRGNN_73907797229592
// MI455X (gfx1250) — matrix operand order repaired and checked by reading, not compiled, not run on hardware
//
#include <hip/hip_runtime.h>
#include <hip/hip_bf16.h>
#include <math.h>

typedef __attribute__((ext_vector_type(16))) _Float16 v16h;
typedef __attribute__((ext_vector_type(8)))  _Float16 v8h;
typedef __attribute__((ext_vector_type(4)))  _Float16 v4h;
typedef __attribute__((ext_vector_type(8)))  float    v8f;

#define HDIM  64
#define VOC   200000
#define BATCH 1024
#define NNODE 50
#define SEQ   50
#define ROWS  (BATCH*NNODE)   // 51200

static __device__ __forceinline__ float sigmoidf_(float x) {
  return 1.0f / (1.0f + __expf(-x));
}

// ---------------------------------------------------------------------------
// Generic WMMA GEMM:  C[M,N] (f32) = A[M,*lda] (f16, first K cols) @ B + bias
// B given transposed: Bt[N,K] row-major, so B[k,n] = Bt[n*K + k].
// REQUIREMENT: M % 16 == 0 (all call sites: 51200 / 51200 / 1024).
// Per-wave tile: 16 rows x 64 cols (4 WMMA n-tiles).
//   K    : compile-time, multiple of 32
//   LDC  : compile-time row stride of C so stores use immediate offsets
//   HASB : bias present
//   NTST : nontemporal stores (logits GEMM: don't pollute L2 with 819MB out)
// Operand layouts per CDNA5 ISA 7.12.2:
//   A (16x32 f16):  lane (m=lane&15, half=lane>>4) holds k = 8*half+(e&7)+16*(e>>3)
//   B (32x16 f16):  lane (n=lane&15, half=lane>>4) holds k = 8*half+(e&7)+16*(e>>3)
//   C/D (16x16 f32): element r -> row m0 + 8*half + r, col n0 + (lane&15)
// ---------------------------------------------------------------------------
template<int K, int LDC, bool HASB, bool NTST>
__global__ __launch_bounds__(256) void k_gemm_wmma(
    const _Float16* __restrict__ A, int lda,
    const _Float16* __restrict__ Bt,
    const float* __restrict__ bias,
    float* __restrict__ C,
    int M, int N)
{
  constexpr int NKT = K / 32;
  const int lane = threadIdx.x & 31;
  const int wave = threadIdx.x >> 5;
  const int tiles_m  = M >> 4;
  const int tiles_ng = (N + 63) >> 6;
  const long long gw = (long long)blockIdx.x * 8 + wave;
  if (gw >= (long long)tiles_m * tiles_ng) return;
  const int tm = (int)(gw % tiles_m);
  const int tg = (int)(gw / tiles_m);
  const int m0 = tm << 4;
  const int n0 = tg << 6;
  const int half = lane >> 4;

  // A operands for all K-tiles (16B-aligned vector loads; lda multiple of 32)
  const _Float16* ap = A + (long long)(m0 + (lane & 15)) * lda + half * 8;
  v16h a[NKT];
#pragma unroll
  for (int kt = 0; kt < NKT; ++kt) {
    v8h lo = *(const v8h*)(ap + kt * 32);
    v8h hi = *(const v8h*)(ap + kt * 32 + 16);
    a[kt] = __builtin_shufflevector(lo, hi, 0,1,2,3,4,5,6,7,8,9,10,11,12,13,14,15);
  }

#pragma unroll
  for (int nt = 0; nt < 4; ++nt) {
    const int n = n0 + nt * 16 + (lane & 15);
    const int ncl = (n < N) ? n : (N - 1);
    const _Float16* bp = Bt + (long long)ncl * K + half * 8;
    // speculative prefetch of the next 64-col group (fault-safe per ISA 10.5)
    __builtin_prefetch(bp + (long long)64 * K, 0, 1);

    // hoist ALL B loads for this n-tile: one clause of 2*NKT b128 loads,
    // a single wait, then a back-to-back WMMA accumulate chain
    v16h b[NKT];
#pragma unroll
    for (int kt = 0; kt < NKT; ++kt) {
      v8h blo = *(const v8h*)(bp + kt * 32);
      v8h bhi = *(const v8h*)(bp + kt * 32 + 16);
      b[kt] = __builtin_shufflevector(blo, bhi, 0,1,2,3,4,5,6,7,8,9,10,11,12,13,14,15);
    }

    v8f acc = {0.f, 0.f, 0.f, 0.f, 0.f, 0.f, 0.f, 0.f};
#pragma unroll
    for (int kt = 0; kt < NKT; ++kt) {
      acc = __builtin_amdgcn_wmma_f32_16x16x32_f16(
          false, a[kt], false, b[kt], (short)0, acc, false, false);
    }

    if (n < N) {
      const float bb = HASB ? bias[n] : 0.f;
      float* cp = C + (long long)(m0 + half * 8) * LDC + n;
#pragma unroll
      for (int r = 0; r < 8; ++r) {
        const float v = acc[r] + bb;
        if (NTST) __builtin_nontemporal_store(v, cp + (long long)r * LDC);
        else      cp[(long long)r * LDC] = v;
      }
    }
  }
}

template<int K, int LDC, bool HASB, bool NTST>
static void launch_gemm(const _Float16* A, int lda, const _Float16* Bt,
                        const float* bias, float* C,
                        int M, int N, hipStream_t stream)
{
  long long tiles = (long long)(M / 16) * ((N + 63) / 64);
  int blocks = (int)((tiles + 7) / 8);
  k_gemm_wmma<K, LDC, HASB, NTST><<<blocks, 256, 0, stream>>>(A, lda, Bt, bias, C, M, N);
}

// ---------------------------------------------------------------------------
// Pack weights into f16 transposed [N,K] layouts + fused biases.
// ---------------------------------------------------------------------------
__global__ void k_pack(const float* __restrict__ ein_w, const float* __restrict__ ein_b,
                       const float* __restrict__ eou_w, const float* __restrict__ eou_b,
                       const float* __restrict__ igate_w, const float* __restrict__ igate_b,
                       const float* __restrict__ hgate_w, const float* __restrict__ hgate_b,
                       const float* __restrict__ fc2_w,
                       _Float16* W1t, float* b1,
                       _Float16* Writ, float* bri,
                       _Float16* Wnit, float* bni,
                       _Float16* Wnht, float* bnh,
                       _Float16* fc2t)
{
  const int id = blockIdx.x * blockDim.x + threadIdx.x;
  if (id < 128 * 64) {   // [ein|eou] -> [128,64]
    int n = id / 64, k = id % 64;
    W1t[id] = (_Float16)(n < 64 ? ein_w[k * 64 + n] : eou_w[k * 64 + (n - 64)]);
  }
  if (id < 128 * 192) {  // stacked r|i gate weights over K=[inputs(128)|h(64)] -> [128,192]
    int n = id / 192, k = id % 192;
    Writ[id] = (_Float16)(k < 128 ? igate_w[k * 192 + n]
                                  : hgate_w[(k - 128) * 192 + n]);
  }
  if (id < 64 * 128) {   // i_n weights -> [64,128]
    int n = id / 128, k = id % 128;
    Wnit[id] = (_Float16)igate_w[k * 192 + 128 + n];
  }
  if (id < 64 * 64) {    // h_n weights -> [64,64] ; fc2 -> [64,64]
    int n = id / 64, k = id % 64;
    Wnht[id] = (_Float16)hgate_w[k * 192 + 128 + n];
    fc2t[id] = (_Float16)fc2_w[k * 64 + n];
  }
  if (id < 128) {
    b1[id]  = (id < 64) ? ein_b[id] : eou_b[id - 64];
    bri[id] = igate_b[id] + hgate_b[id];
  }
  if (id < 64) {
    bni[id] = igate_b[128 + id];
    bnh[id] = hgate_b[128 + id];
  }
}

// emb f32 -> f16 (streams 51MB -> 25.6MB once; emb_h then lives in L2)
__global__ void k_emb2h(const float* __restrict__ emb, _Float16* __restrict__ emb_h)
{
  long long id = (long long)blockIdx.x * blockDim.x + threadIdx.x;
  if (id >= (long long)VOC * HDIM / 4) return;
  float4 f = ((const float4*)emb)[id];
  v4h o = {(_Float16)f.x, (_Float16)f.y, (_Float16)f.z, (_Float16)f.w};
  ((v4h*)emb_h)[id] = o;
}

// h = emb[items] gathered (f16) into X columns 128..191 (X row stride 192)
__global__ void k_gather_h(const float* __restrict__ emb, const int* __restrict__ items,
                           _Float16* __restrict__ X)
{
  int id = blockIdx.x * 256 + threadIdx.x;
  if (id >= ROWS * HDIM) return;
  int r = id >> 6, j = id & 63;
  X[(long long)r * 192 + 128 + j] = (_Float16)emb[(long long)items[r] * HDIM + j];
}

// Per-batch adjacency matmuls, staged in LDS:
//   X[:,0:64]  = ain @ hin + b_iah ;  X[:,64:128] = aou @ hou + b_oah
__global__ __launch_bounds__(256) void k_adj(
    const float* __restrict__ C1, const float* __restrict__ ain,
    const float* __restrict__ aou, const float* __restrict__ b_iah,
    const float* __restrict__ b_oah, _Float16* __restrict__ X)
{
  __shared__ float sh[NNODE * 128];
  __shared__ float sa[NNODE * NNODE];
  __shared__ float so[NNODE * NNODE];
  const int b = blockIdx.x, tid = threadIdx.x;
  for (int i = tid; i < NNODE * 128; i += 256)
    sh[i] = C1[(long long)b * NNODE * 128 + i];
  for (int i = tid; i < NNODE * NNODE; i += 256) {
    sa[i] = ain[(long long)b * NNODE * NNODE + i];
    so[i] = aou[(long long)b * NNODE * NNODE + i];
  }
  __syncthreads();
  for (int o = tid; o < NNODE * 128; o += 256) {
    const int n = o >> 7, j = o & 127;
    float acc;
    if (j < 64) {
      acc = b_iah[j];
      const float* ar = &sa[n * NNODE];
      for (int m = 0; m < NNODE; ++m) acc += ar[m] * sh[m * 128 + j];
    } else {
      acc = b_oah[j - 64];
      const float* ar = &so[n * NNODE];
      for (int m = 0; m < NNODE; ++m) acc += ar[m] * sh[m * 128 + j];
    }
    X[((long long)b * NNODE + n) * 192 + j] = (_Float16)acc;
  }
}

// GRU pointwise update -> hnew (f32)
__global__ void k_gru(const float* __restrict__ Gri, const float* __restrict__ INn,
                      const float* __restrict__ HNn, const float* __restrict__ emb,
                      const int* __restrict__ items, float* __restrict__ hnew)
{
  int id = blockIdx.x * 256 + threadIdx.x;
  if (id >= ROWS * HDIM) return;
  int r = id >> 6, j = id & 63;
  float gr = Gri[(long long)r * 128 + j];        // i_r + h_r (+biases)
  float gi = Gri[(long long)r * 128 + 64 + j];   // i_i + h_i (+biases)
  float in_n = INn[id];
  float h_n  = HNn[id];
  float h = emb[(long long)items[r] * HDIM + j];
  float reset = sigmoidf_(gr);
  float ig    = sigmoidf_(gi);
  float ng    = tanhf(in_n + reset * h_n);
  hnew[id] = ng + ig * (h - ng);
}

// seq_h (f16) = hnew gathered to sequence order (for q2 GEMM)
__global__ void k_seqg(const float* __restrict__ hnew, const int* __restrict__ alias_,
                       _Float16* __restrict__ seq_h)
{
  int id = blockIdx.x * 256 + threadIdx.x;
  if (id >= BATCH * SEQ * HDIM) return;
  int row = id >> 6, j = id & 63;
  int b = row / SEQ, s = row % SEQ;
  int node = alias_[b * SEQ + s];
  seq_h[id] = (_Float16)hnew[((long long)b * NNODE + node) * HDIM + j];
}

// q1 = ht @ fc1_w + fc1_b   (ht = seq[b, sum(mask)-1])
__global__ __launch_bounds__(64) void k_q1(
    const float* __restrict__ hnew, const int* __restrict__ alias_,
    const int* __restrict__ mask, const float* __restrict__ fc1_w,
    const float* __restrict__ fc1_b, float* __restrict__ q1)
{
  __shared__ float ht[64];
  const int b = blockIdx.x, j = threadIdx.x;
  int last = -1;
  for (int s = 0; s < SEQ; ++s) last += mask[b * SEQ + s];
  if (last < 0) last = 0;
  int node = alias_[b * SEQ + last];
  ht[j] = hnew[((long long)b * NNODE + node) * HDIM + j];
  __syncthreads();
  float acc = fc1_b[j];
  for (int k = 0; k < 64; ++k) acc += ht[k] * fc1_w[k * 64 + j];
  q1[b * 64 + j] = acc;
}

// alpha + weighted readout -> a (written as f16 for logits GEMM)
__global__ __launch_bounds__(64) void k_attn(
    const float* __restrict__ q1, const float* __restrict__ q2,
    const float* __restrict__ hnew, const int* __restrict__ alias_,
    const int* __restrict__ mask, const float* __restrict__ fc3_w,
    _Float16* __restrict__ a_h)
{
  __shared__ float sq1[64], sf3[64], sal[SEQ];
  const int b = blockIdx.x, t = threadIdx.x;
  sq1[t] = q1[b * 64 + t];
  sf3[t] = fc3_w[t];
  __syncthreads();
  if (t < SEQ) {
    float acc = 0.f;
    const float* q2r = q2 + ((long long)b * SEQ + t) * 64;
    for (int j = 0; j < 64; ++j) acc += sigmoidf_(sq1[j] + q2r[j]) * sf3[j];
    sal[t] = acc * (float)mask[b * SEQ + t];
  }
  __syncthreads();
  float acc = 0.f;
  for (int s = 0; s < SEQ; ++s) {
    int node = alias_[b * SEQ + s];
    acc += sal[s] * hnew[((long long)b * NNODE + node) * HDIM + t];
  }
  a_h[b * 64 + t] = (_Float16)acc;
}

// ---------------------------------------------------------------------------
extern "C" void kernel_launch(void* const* d_in, const int* in_sizes, int n_in,
                              void* d_out, int out_size, void* d_ws, size_t ws_size,
                              hipStream_t stream)
{
  (void)in_sizes; (void)n_in; (void)out_size; (void)ws_size;

  const int*   alias_  = (const int*)  d_in[0];
  const float* ain     = (const float*)d_in[1];
  const float* aou     = (const float*)d_in[2];
  const int*   items   = (const int*)  d_in[3];
  const int*   mask    = (const int*)  d_in[4];
  /* d_in[5] = edge_index (unused) */
  const float* emb     = (const float*)d_in[6];
  const float* ein_w   = (const float*)d_in[7];
  const float* ein_b   = (const float*)d_in[8];
  const float* eou_w   = (const float*)d_in[9];
  const float* eou_b   = (const float*)d_in[10];
  const float* b_iah   = (const float*)d_in[11];
  const float* b_oah   = (const float*)d_in[12];
  const float* igate_w = (const float*)d_in[13];
  const float* igate_b = (const float*)d_in[14];
  const float* hgate_w = (const float*)d_in[15];
  const float* hgate_b = (const float*)d_in[16];
  const float* fc1_w   = (const float*)d_in[17];
  const float* fc1_b   = (const float*)d_in[18];
  const float* fc2_w   = (const float*)d_in[19];
  const float* fc2_b   = (const float*)d_in[20];
  const float* fc3_w   = (const float*)d_in[21];

  // ---- workspace carve-out (256B aligned), with lifetime-based aliasing ----
  char* ws = (char*)d_ws;
  size_t off = 0;
  auto take = [&](size_t bytes) -> char* {
    char* p = ws + off;
    off = (off + bytes + 255) & ~(size_t)255;
    return p;
  };
  _Float16* emb_h = (_Float16*)take((size_t)VOC * HDIM * 2);          // 25.6 MB (live to end)
  float*    C1    = (float*)   take((size_t)ROWS * 128 * 4);          // 26.2 MB  [hin|hou]
  _Float16* X     = (_Float16*)take((size_t)ROWS * 192 * 2);          // 19.7 MB  [iin|iou|h]
  float*    INb   = (float*)   take((size_t)ROWS * HDIM * 4);         // 13.1 MB  i_n
  float*    HNb   = (float*)   take((size_t)ROWS * HDIM * 4);         // 13.1 MB  h_n
  float*    q1    = (float*)   take((size_t)BATCH * HDIM * 4);
  _Float16* a_h   = (_Float16*)take((size_t)BATCH * HDIM * 2);
  _Float16* W1t   = (_Float16*)take(128 * 64 * 2);
  _Float16* Writ  = (_Float16*)take(128 * 192 * 2);
  _Float16* Wnit  = (_Float16*)take(64 * 128 * 2);
  _Float16* Wnht  = (_Float16*)take(64 * 64 * 2);
  _Float16* fc2t  = (_Float16*)take(64 * 64 * 2);
  float*    b1    = (float*)   take(128 * 4);
  float*    bri   = (float*)   take(128 * 4);
  float*    bni   = (float*)   take(64 * 4);
  float*    bnh   = (float*)   take(64 * 4);
  // aliases (the aliased buffer is dead by the time the alias is written):
  float*    Gri   = C1;            // r|i gate preacts overwrite C1 after k_adj
  float*    hnew  = (float*)X;     // GRU output overwrites X after gate GEMMs
  float*    q2    = INb;           // q2 overwrites i_n after k_gru
  _Float16* seq_h = (_Float16*)HNb;// seq gather overwrites h_n after k_gru

  const int ELEMS = ROWS * HDIM;           // 3,276,800
  const int GBLK  = (ELEMS + 255) / 256;

  // 1) weight packing + emb f32->f16
  k_pack<<<96, 256, 0, stream>>>(ein_w, ein_b, eou_w, eou_b, igate_w, igate_b,
                                 hgate_w, hgate_b, fc2_w,
                                 W1t, b1, Writ, bri, Wnit, bni, Wnht, bnh, fc2t);
  {
    long long tot4 = (long long)VOC * HDIM / 4;
    k_emb2h<<<(int)((tot4 + 255) / 256), 256, 0, stream>>>(emb, emb_h);
  }
  // 2) h = emb[items] -> X[:,128:192]
  k_gather_h<<<GBLK, 256, 0, stream>>>(emb, items, X);
  // 3) C1 = h @ [ein_w|eou_w] + [ein_b|eou_b]       (WMMA, K=64, N=128)
  launch_gemm<64, 128, true, false>(X + 128, 192, W1t, b1, C1, ROWS, 128, stream);
  // 4) X[:,0:128] = [ain@hin + b_iah | aou@hou + b_oah]
  k_adj<<<BATCH, 256, 0, stream>>>(C1, ain, aou, b_iah, b_oah, X);
  // 5) fused r|i gate preactivations: Gri = X @ Writ + (ib+hb)   (K=192, N=128)
  launch_gemm<192, 128, true, false>(X, 192, Writ, bri, Gri, ROWS, 128, stream);
  // 6) i_n = inputs @ igate_w[:,128:] + b              (K=128, N=64)
  launch_gemm<128, 64, true, false>(X, 192, Wnit, bni, INb, ROWS, 64, stream);
  // 7) h_n = h @ hgate_w[:,128:] + b                   (K=64, N=64)
  launch_gemm<64, 64, true, false>(X + 128, 192, Wnht, bnh, HNb, ROWS, 64, stream);
  // 8) GRU pointwise -> hnew
  k_gru<<<GBLK, 256, 0, stream>>>(Gri, INb, HNb, emb, items, hnew);
  // 9) seq gather (f16)
  k_seqg<<<GBLK, 256, 0, stream>>>(hnew, alias_, seq_h);
  // 10) q2 = seq @ fc2_w + fc2_b                       (K=64, N=64)
  launch_gemm<64, 64, true, false>(seq_h, 64, fc2t, fc2_b, q2, BATCH * SEQ, 64, stream);
  // 11) q1 = ht @ fc1_w + fc1_b
  k_q1<<<BATCH, 64, 0, stream>>>(hnew, alias_, mask, fc1_w, fc1_b, q1);
  // 12) attention readout -> a (f16)
  k_attn<<<BATCH, 64, 0, stream>>>(q1, q2, hnew, alias_, mask, fc3_w, a_h);
  // 13) logits = a @ emb[1:].T -> d_out [1024, 199999]; nontemporal stores
  launch_gemm<64, VOC - 1, false, true>(a_h, 64, emb_h + HDIM, nullptr,
                                        (float*)d_out, BATCH, VOC - 1, stream);
}
